// HAN_37426345017738
// MI455X (gfx1250) — compile-verified, matrix operand order checked by reading, not run on hardware
//
#include <hip/hip_runtime.h>
#include <hip/hip_bf16.h>

// ---------------------------------------------------------------------------
// HAN forward for MI455X (gfx1250, wave32, WMMA).
// GEMMs: v_wmma_f32_16x16x32_bf16 (f32 accum, bf16 operands).
// Weight matrices are pre-converted+pre-swizzled once into WMMA B-fragment
// order (bf16), so each B fragment is one 32-byte coalesced load, no cvt.
// Memory-bound problem (~2 GB traffic @ 23.3 TB/s); GEMM FLOPs are trivial.
// ---------------------------------------------------------------------------

typedef __attribute__((ext_vector_type(16))) __bf16 v16bf;
typedef __attribute__((ext_vector_type(8)))  float  v8f;

#define DD 128   // embedding dim (K of every GEMM)
#define CC 64    // classifier output dim

__device__ __forceinline__ float wave_sum(float v) {
#pragma unroll
  for (int off = 16; off > 0; off >>= 1) v += __shfl_xor(v, off, 32);
  return v;
}

// order-preserving float <-> uint encoding for atomicMax-based segment max
__device__ __forceinline__ unsigned encf(float f) {
  unsigned u = __float_as_uint(f);
  return (u & 0x80000000u) ? ~u : (u | 0x80000000u);
}
__device__ __forceinline__ float decf(unsigned u) {
  return (u & 0x80000000u) ? __uint_as_float(u ^ 0x80000000u)
                           : __uint_as_float(~u);
}

// ---------------------------------------------------------------------------
// Weight prep: B[128,NB] f32 row-major  ->  Bt bf16 in WMMA fragment order.
// Linear index t = (((ct*4 + kc)*32 + lane)*16 + i):
//   k = kc*32 + (lane>>4)*16 + i ,  n = ct*16 + (lane&15)
// so each wave's (ct,kc) B-fragment is 1KB contiguous, 32B per lane.
// ---------------------------------------------------------------------------
__global__ __launch_bounds__(256) void k_swizzleB(
    const float* __restrict__ B, __bf16* __restrict__ Bt, int NB)
{
  const int t = blockIdx.x * 256 + threadIdx.x;
  if (t >= NB * DD) return;
  const int i  = t & 15;
  const int l  = (t >> 4) & 31;
  const int kc = (t >> 9) & 3;
  const int ct = t >> 11;
  const int k  = kc * 32 + ((l >> 4) << 4) + i;
  const int n  = ct * 16 + (l & 15);
  Bt[t] = (__bf16)B[(size_t)k * NB + n];
}

__device__ __forceinline__ v16bf load_bfrag(const __bf16* __restrict__ Bt,
                                            int coltile, int kc, int lane) {
  return *(const v16bf*)(Bt + ((((size_t)coltile * 4 + kc) * 32 + lane) << 4));
}

__device__ __forceinline__ v16bf load_afrag_f32(const float* __restrict__ ap) {
  v16bf a;
#pragma unroll
  for (int i = 0; i < 16; ++i) a[i] = (__bf16)ap[i];
  return a;
}

// ---------------------------------------------------------------------------
// out[M,128] = A[M,128] @ B[128,128] + bias      (M multiple of 16)
// block = 256 threads = 8 waves; block tile = 16 rows x 128 cols.
// ---------------------------------------------------------------------------
__global__ __launch_bounds__(256) void k_gemm128_bias(
    const float* __restrict__ A, const __bf16* __restrict__ Bt,
    const float* __restrict__ bias, float* __restrict__ out)
{
  const int lane = threadIdx.x & 31;
  const int wave = threadIdx.x >> 5;
  const int r0   = blockIdx.x << 4;
  const int row  = lane & 15;
  const int kh   = (lane >> 4) << 4;       // K half: 0 or 16
  const int col  = (wave << 4) + row;

  const float* arow = A + (size_t)(r0 + row) * DD + kh;
  v8f acc = {};
#pragma unroll
  for (int kc = 0; kc < 4; ++kc) {
    const v16bf a = load_afrag_f32(arow + kc * 32);
    const v16bf b = load_bfrag(Bt, wave, kc, lane);
    acc = __builtin_amdgcn_wmma_f32_16x16x32_bf16(false, a, false, b,
                                                  (short)0, acc, false, false);
  }
  const float bv = bias[col];
  const int   mb = r0 + ((lane >> 4) << 3);
#pragma unroll
  for (int r = 0; r < 8; ++r)
    out[(size_t)(mb + r) * DD + col] = acc[r] + bv;
}

// ---------------------------------------------------------------------------
// semantic score: atomically accumulate sum_{n,d} q[d]*tanh((A@Kw + kb)[n,d])
// (divided by N later, in the softmax kernel)
// ---------------------------------------------------------------------------
__global__ __launch_bounds__(256) void k_gemm128_tanh_score(
    const float* __restrict__ A, const __bf16* __restrict__ Bt,
    const float* __restrict__ bias, const float* __restrict__ q,
    float* __restrict__ score)
{
  const int lane = threadIdx.x & 31;
  const int wave = threadIdx.x >> 5;
  const int r0   = blockIdx.x << 4;
  const int row  = lane & 15;
  const int kh   = (lane >> 4) << 4;
  const int col  = (wave << 4) + row;

  const float* arow = A + (size_t)(r0 + row) * DD + kh;
  v8f acc = {};
#pragma unroll
  for (int kc = 0; kc < 4; ++kc) {
    const v16bf a = load_afrag_f32(arow + kc * 32);
    const v16bf b = load_bfrag(Bt, wave, kc, lane);
    acc = __builtin_amdgcn_wmma_f32_16x16x32_bf16(false, a, false, b,
                                                  (short)0, acc, false, false);
  }
  const float bv = bias[col];
  const float qv = q[col];
  float partial = 0.f;
#pragma unroll
  for (int r = 0; r < 8; ++r) partial += tanhf(acc[r] + bv) * qv;
  partial = wave_sum(partial);
  if (lane == 0) atomicAdd(score, partial);
}

// ---------------------------------------------------------------------------
// classifier: out[M,64] = (w0*A0 + w1*A1)[M,128] @ B[128,64] + bias
// A1==nullptr -> plain A0. wts==nullptr -> (1,0).
// block = 256 threads = 8 waves; tile = 32 rows x 64 cols (2 row-tiles).
// ---------------------------------------------------------------------------
__global__ __launch_bounds__(256) void k_gemm64_cls(
    const float* __restrict__ A0, const float* __restrict__ A1,
    const float* __restrict__ wts,
    const __bf16* __restrict__ Bt, const float* __restrict__ bias,
    float* __restrict__ out, int M)
{
  const int lane = threadIdx.x & 31;
  const int wave = threadIdx.x >> 5;
  const int r0   = blockIdx.x * 32 + ((wave >> 2) << 4);
  if (r0 >= M) return;                      // wave-uniform: EXEC stays all-1s
  const int ct   = wave & 3;
  const int row  = lane & 15;
  const int kh   = (lane >> 4) << 4;
  const int col  = (ct << 4) + row;

  float w0 = 1.f, w1 = 0.f;
  if (wts) { w0 = wts[0]; w1 = wts[1]; }

  const float* a0row = A0 + (size_t)(r0 + row) * DD + kh;
  const float* a1row = A1 ? A1 + (size_t)(r0 + row) * DD + kh : nullptr;
  v8f acc = {};
#pragma unroll
  for (int kc = 0; kc < 4; ++kc) {
    v16bf a;
    if (A1) {
      const float* p0 = a0row + kc * 32;
      const float* p1 = a1row + kc * 32;
#pragma unroll
      for (int i = 0; i < 16; ++i) a[i] = (__bf16)(w0 * p0[i] + w1 * p1[i]);
    } else {
      a = load_afrag_f32(a0row + kc * 32);
    }
    const v16bf b = load_bfrag(Bt, ct, kc, lane);
    acc = __builtin_amdgcn_wmma_f32_16x16x32_bf16(false, a, false, b,
                                                  (short)0, acc, false, false);
  }
  const float bv = bias[col];
  const int   mb = r0 + ((lane >> 4) << 3);
#pragma unroll
  for (int r = 0; r < 8; ++r)
    out[(size_t)(mb + r) * CC + col] = acc[r] + bv;
}

// ---------------------------------------------------------------------------
// s[n] = X[n,:] . a      one wave per row (float4 per lane -> wave reduce)
// ---------------------------------------------------------------------------
__global__ __launch_bounds__(256) void k_rowdot(
    const float* __restrict__ X, const float* __restrict__ a,
    float* __restrict__ s, int N)
{
  const int row = blockIdx.x * 8 + (threadIdx.x >> 5);
  if (row >= N) return;
  const int lane = threadIdx.x & 31;
  const float4 x  = ((const float4*)(X + (size_t)row * DD))[lane];
  const float4 av = ((const float4*)a)[lane];
  float p = x.x * av.x + x.y * av.y + x.z * av.z + x.w * av.w;
  p = wave_sum(p);
  if (lane == 0) s[row] = p;
}

// alpha = leaky_relu(s_src[src] + s_dst[dst]); segment max via encoded atomicMax
__global__ __launch_bounds__(256) void k_edge_alpha_max(
    const int* __restrict__ src, const int* __restrict__ dst,
    const float* __restrict__ ssrc, const float* __restrict__ sdst,
    float* __restrict__ alpha, unsigned* __restrict__ menc, int nE)
{
  const int e = blockIdx.x * 256 + threadIdx.x;
  if (e >= nE) return;
  float al = ssrc[src[e]] + sdst[dst[e]];
  al = (al >= 0.f) ? al : 0.2f * al;
  alpha[e] = al;
  atomicMax(&menc[dst[e]], encf(al));
}

// ex = exp(alpha - m[dst]) (in place over alpha); den[dst] += ex
__global__ __launch_bounds__(256) void k_edge_exp_den(
    const int* __restrict__ dst, const unsigned* __restrict__ menc,
    float* __restrict__ alpha_ex, float* __restrict__ den, int nE)
{
  const int e = blockIdx.x * 256 + threadIdx.x;
  if (e >= nE) return;
  const int d = dst[e];
  const float ex = __expf(alpha_ex[e] - decf(menc[d]));
  alpha_ex[e] = ex;
  atomicAdd(&den[d], ex);
}

// out[dst] += x_src[src] * (ex/(den+1e-16))   — one wave per edge, float4/lane
__global__ __launch_bounds__(256) void k_edge_scatter(
    const int* __restrict__ src, const int* __restrict__ dst,
    const float* __restrict__ ex, const float* __restrict__ den,
    const float* __restrict__ xsrc, float* __restrict__ out, int nE)
{
  const int e = blockIdx.x * 8 + (threadIdx.x >> 5);
  if (e >= nE) return;
  const int lane = threadIdx.x & 31;
  const int s = src[e], d = dst[e];
  const float w = ex[e] / (den[d] + 1e-16f);
  const float4 x = ((const float4*)(xsrc + (size_t)s * DD))[lane];
  float* o = out + (size_t)d * DD + lane * 4;
  atomicAdd(o + 0, x.x * w);
  atomicAdd(o + 1, x.y * w);
  atomicAdd(o + 2, x.z * w);
  atomicAdd(o + 3, x.w * w);
}

__global__ __launch_bounds__(256) void k_relu(float* __restrict__ p, size_t n)
{
  size_t i = (size_t)blockIdx.x * blockDim.x + threadIdx.x;
  const size_t st = (size_t)gridDim.x * blockDim.x;
  for (; i < n; i += st) p[i] = fmaxf(p[i], 0.f);
}

// wts = softmax([s0,s1]/N)   (single thread; tiny)
__global__ void k_softmax2(const float* __restrict__ scores,
                           float* __restrict__ wts, float invN)
{
  const float s0 = scores[0] * invN, s1 = scores[1] * invN;
  const float m  = fmaxf(s0, s1);
  const float e0 = __expf(s0 - m), e1 = __expf(s1 - m);
  const float inv = 1.f / (e0 + e1);
  wts[0] = e0 * inv;
  wts[1] = e1 * inv;
}

// ---------------------------------------------------------------------------
extern "C" void kernel_launch(void* const* d_in, const int* in_sizes, int n_in,
                              void* d_out, int out_size, void* d_ws, size_t ws_size,
                              hipStream_t stream)
{
  const int U = in_sizes[0];   // 200000
  const int L = in_sizes[1];   // 100000
  const int T = in_sizes[2];   // 50000
  const int E = in_sizes[3];   // 300000

  const int* e_lu_src = (const int*)d_in[3];
  const int* e_lu_dst = (const int*)d_in[4];
  const int* e_tu_src = (const int*)d_in[5];
  const int* e_tu_dst = (const int*)d_in[6];
  const int* e_ul_src = (const int*)d_in[7];
  const int* e_ul_dst = (const int*)d_in[8];
  const int* e_ut_src = (const int*)d_in[9];
  const int* e_ut_dst = (const int*)d_in[10];
  const float* emb_user = (const float*)d_in[11];
  const float* emb_loc  = (const float*)d_in[12];
  const float* emb_time = (const float*)d_in[13];
  const float* pu_w = (const float*)d_in[14]; const float* pu_b = (const float*)d_in[15];
  const float* pl_w = (const float*)d_in[16]; const float* pl_b = (const float*)d_in[17];
  const float* pt_w = (const float*)d_in[18]; const float* pt_b = (const float*)d_in[19];
  const float* a_lu_s = (const float*)d_in[20]; const float* a_lu_d = (const float*)d_in[21];
  const float* a_tu_s = (const float*)d_in[22]; const float* a_tu_d = (const float*)d_in[23];
  const float* a_ul_s = (const float*)d_in[24]; const float* a_ul_d = (const float*)d_in[25];
  const float* a_ut_s = (const float*)d_in[26]; const float* a_ut_d = (const float*)d_in[27];
  const float* k_w = (const float*)d_in[28]; const float* k_b = (const float*)d_in[29];
  const float* q_v = (const float*)d_in[30];
  const float* cu_w = (const float*)d_in[31]; const float* cu_b = (const float*)d_in[32];
  const float* ct_w = (const float*)d_in[33]; const float* ct_b = (const float*)d_in[34];
  const float* cl_w = (const float*)d_in[35]; const float* cl_b = (const float*)d_in[36];

  // ---- workspace layout (floats) ----
  float* ws = (float*)d_ws;
  size_t off = 0;
  auto alloc = [&](size_t n) { float* p = ws + off; off += n; return p; };

  float* xu = alloc((size_t)U * DD);
  float* xl = alloc((size_t)L * DD);
  float* xt = alloc((size_t)T * DD);
  // ---- zero-init region (contiguous, one memset) ----
  float* zero_base = ws + off;
  float* o_lu = alloc((size_t)U * DD);
  float* o_tu = alloc((size_t)U * DD);
  float* o_ul = alloc((size_t)L * DD);
  float* o_ut = alloc((size_t)T * DD);
  float* den_lu = alloc(U);
  float* den_tu = alloc(U);
  float* den_ul = alloc(L);
  float* den_ut = alloc(T);
  unsigned* m_lu = (unsigned*)alloc(U);
  unsigned* m_tu = (unsigned*)alloc(U);
  unsigned* m_ul = (unsigned*)alloc(L);
  unsigned* m_ut = (unsigned*)alloc(T);
  float* scores = alloc(2);
  float* semwts = alloc(2);
  const size_t zero_floats = (size_t)(ws + off - zero_base);
  // ---- non-zeroed scratch ----
  float* s_lu_src = alloc(L); float* s_lu_dst = alloc(U);
  float* s_tu_src = alloc(T); float* s_tu_dst = alloc(U);
  float* s_ul_src = alloc(U); float* s_ul_dst = alloc(L);
  float* s_ut_src = alloc(U); float* s_ut_dst = alloc(T);
  float* al_lu = alloc(E); float* al_tu = alloc(E);
  float* al_ul = alloc(E); float* al_ut = alloc(E);
  // bf16 fragment-ordered weights (sizes are multiples of 2 floats)
  __bf16* bt_pu = (__bf16*)alloc(DD * DD / 2);
  __bf16* bt_pl = (__bf16*)alloc(DD * DD / 2);
  __bf16* bt_pt = (__bf16*)alloc(DD * DD / 2);
  __bf16* bt_kw = (__bf16*)alloc(DD * DD / 2);
  __bf16* bt_cu = (__bf16*)alloc(DD * CC / 2);
  __bf16* bt_ct = (__bf16*)alloc(DD * CC / 2);
  __bf16* bt_cl = (__bf16*)alloc(DD * CC / 2);
  (void)ws_size; (void)n_in; (void)out_size;

  hipMemsetAsync(zero_base, 0, zero_floats * sizeof(float), stream);

  // ---- weight prep: f32 row-major -> bf16 WMMA-fragment order ----
  const int g128 = (DD * DD + 255) / 256, g64 = (DD * CC + 255) / 256;
  k_swizzleB<<<g128, 256, 0, stream>>>(pu_w, bt_pu, DD);
  k_swizzleB<<<g128, 256, 0, stream>>>(pl_w, bt_pl, DD);
  k_swizzleB<<<g128, 256, 0, stream>>>(pt_w, bt_pt, DD);
  k_swizzleB<<<g128, 256, 0, stream>>>(k_w,  bt_kw, DD);
  k_swizzleB<<<g64,  256, 0, stream>>>(cu_w, bt_cu, CC);
  k_swizzleB<<<g64,  256, 0, stream>>>(ct_w, bt_ct, CC);
  k_swizzleB<<<g64,  256, 0, stream>>>(cl_w, bt_cl, CC);

  // ---- node projections (WMMA GEMMs); user_idx/loc_idx/time_idx are arange ----
  k_gemm128_bias<<<U / 16, 256, 0, stream>>>(emb_user, bt_pu, pu_b, xu);
  k_gemm128_bias<<<L / 16, 256, 0, stream>>>(emb_loc,  bt_pl, pl_b, xl);
  k_gemm128_bias<<<T / 16, 256, 0, stream>>>(emb_time, bt_pt, pt_b, xt);

  // ---- per-node attention dots ----
  k_rowdot<<<(L + 7) / 8, 256, 0, stream>>>(xl, a_lu_s, s_lu_src, L);
  k_rowdot<<<(U + 7) / 8, 256, 0, stream>>>(xu, a_lu_d, s_lu_dst, U);
  k_rowdot<<<(T + 7) / 8, 256, 0, stream>>>(xt, a_tu_s, s_tu_src, T);
  k_rowdot<<<(U + 7) / 8, 256, 0, stream>>>(xu, a_tu_d, s_tu_dst, U);
  k_rowdot<<<(U + 7) / 8, 256, 0, stream>>>(xu, a_ul_s, s_ul_src, U);
  k_rowdot<<<(L + 7) / 8, 256, 0, stream>>>(xl, a_ul_d, s_ul_dst, L);
  k_rowdot<<<(U + 7) / 8, 256, 0, stream>>>(xu, a_ut_s, s_ut_src, U);
  k_rowdot<<<(T + 7) / 8, 256, 0, stream>>>(xt, a_ut_d, s_ut_dst, T);

  // ---- edge attention: alpha/leaky -> segmax -> exp/segsum -> scatter ----
  const int gE = (E + 255) / 256, gS = (E + 7) / 8;
  // loc -> user
  k_edge_alpha_max<<<gE, 256, 0, stream>>>(e_lu_src, e_lu_dst, s_lu_src, s_lu_dst, al_lu, m_lu, E);
  k_edge_exp_den <<<gE, 256, 0, stream>>>(e_lu_dst, m_lu, al_lu, den_lu, E);
  k_edge_scatter <<<gS, 256, 0, stream>>>(e_lu_src, e_lu_dst, al_lu, den_lu, xl, o_lu, E);
  // time -> user
  k_edge_alpha_max<<<gE, 256, 0, stream>>>(e_tu_src, e_tu_dst, s_tu_src, s_tu_dst, al_tu, m_tu, E);
  k_edge_exp_den <<<gE, 256, 0, stream>>>(e_tu_dst, m_tu, al_tu, den_tu, E);
  k_edge_scatter <<<gS, 256, 0, stream>>>(e_tu_src, e_tu_dst, al_tu, den_tu, xt, o_tu, E);
  // user -> loc
  k_edge_alpha_max<<<gE, 256, 0, stream>>>(e_ul_src, e_ul_dst, s_ul_src, s_ul_dst, al_ul, m_ul, E);
  k_edge_exp_den <<<gE, 256, 0, stream>>>(e_ul_dst, m_ul, al_ul, den_ul, E);
  k_edge_scatter <<<gS, 256, 0, stream>>>(e_ul_src, e_ul_dst, al_ul, den_ul, xu, o_ul, E);
  // user -> time
  k_edge_alpha_max<<<gE, 256, 0, stream>>>(e_ut_src, e_ut_dst, s_ut_src, s_ut_dst, al_ut, m_ut, E);
  k_edge_exp_den <<<gE, 256, 0, stream>>>(e_ut_dst, m_ut, al_ut, den_ut, E);
  k_edge_scatter <<<gS, 256, 0, stream>>>(e_ut_src, e_ut_dst, al_ut, den_ut, xu, o_ut, E);

  // ---- relu over all 4 contiguous metapath outputs in one pass ----
  const size_t n_relu = (size_t)(2 * (size_t)U + L + T) * DD;
  k_relu<<<8192, 256, 0, stream>>>(o_lu, n_relu);

  // ---- semantic attention: only the user group has >1 metapath.
  //      (softmax over a single score == 1, so hl=o_ul, ht=o_ut exactly.) ----
  k_gemm128_tanh_score<<<U / 16, 256, 0, stream>>>(o_lu, bt_kw, k_b, q_v, scores + 0);
  k_gemm128_tanh_score<<<U / 16, 256, 0, stream>>>(o_tu, bt_kw, k_b, q_v, scores + 1);
  k_softmax2<<<1, 1, 0, stream>>>(scores, semwts, 1.f / (float)U);

  // ---- classifier heads (output order: user, time, loc) ----
  float* out_user = (float*)d_out;
  float* out_time = out_user + (size_t)U * CC;
  float* out_loc  = out_time + (size_t)T * CC;
  k_gemm64_cls<<<(U + 31) / 32, 256, 0, stream>>>(o_lu, o_tu, semwts, bt_cu, cu_b, out_user, U);
  k_gemm64_cls<<<(T + 31) / 32, 256, 0, stream>>>(o_ut, nullptr, nullptr, bt_ct, ct_b, out_time, T);
  k_gemm64_cls<<<(L + 31) / 32, 256, 0, stream>>>(o_ul, nullptr, nullptr, bt_cl, cl_b, out_loc, L);
}
